// StateSpaceExpert_85126251807087
// MI455X (gfx1250) — matrix operand order repaired and checked by reading, not compiled, not run on hardware
//
#include <hip/hip_runtime.h>
#include <hip/hip_bf16.h>
#include <stdint.h>

// ---------------- types ----------------
typedef __attribute__((ext_vector_type(16))) __bf16 v16bf;
typedef __attribute__((ext_vector_type(8)))  __bf16 v8bf;
typedef __attribute__((ext_vector_type(4)))  __bf16 v4bf;
typedef __attribute__((ext_vector_type(8)))  float  v8f;

#define D_MODEL 2048
#define BATCH   32
#define SEQ     512

// round-to-nearest-even f32 -> bf16 (bit-exact, no reliance on cast codegen)
__device__ __forceinline__ __bf16 to_bf16(float f) {
    union { float f; uint32_t u; } in; in.f = f;
    uint32_t r = in.u + 0x7FFFu + ((in.u >> 16) & 1u);
    uint16_t h = (uint16_t)(r >> 16);
    union { uint16_t u; __bf16 b; } out; out.u = h;
    return out.b;
}

// ---------------- f32 -> bf16 convert (4 elems/thread) ----------------
__global__ __launch_bounds__(256)
void cvt_f32_to_bf16(const float* __restrict__ src, __bf16* __restrict__ dst, int n4) {
    int i = blockIdx.x * 256 + threadIdx.x;
    if (i >= n4) return;
    const float4 f = ((const float4*)src)[i];
    v4bf o;
    o[0] = to_bf16(f.x); o[1] = to_bf16(f.y);
    o[2] = to_bf16(f.z); o[3] = to_bf16(f.w);
    ((v4bf*)dst)[i] = o;
}

// ---------------- bf16 WMMA GEMM:  C[M,N] = A[M,K] * W[N,K]^T + bias[N] ----------------
// Block: 256 threads = 8 waves.  Block tile 128(M) x 256(N).
// Wave grid 2(M) x 4(N); wave tile 64 x 64 = 4x4 WMMA 16x16 accumulators.
// 16 WMMAs per K-step of 32 against 8 fragment loads -> 1.0 b128-loads/WMMA.
union Frag { v16bf v; v8bf h[2]; };

__global__ __launch_bounds__(256)
void gemm_bf16_wmma(const __bf16* __restrict__ A,   // M x K row-major
                    const __bf16* __restrict__ W,   // N x K row-major (B = W^T)
                    const float*  __restrict__ bias,// N
                    float*        __restrict__ C,   // M x N row-major, f32
                    int M, int N, int K)
{
    const int lane = threadIdx.x & 31;
    const int wave = threadIdx.x >> 5;
    const int wm = wave & 1;            // wave position along M (0..1)
    const int wn = wave >> 1;           // wave position along N (0..3)
    const int m0 = blockIdx.y * 128 + wm * 64;
    const int n0 = blockIdx.x * 256 + wn * 64;

    const int r = lane & 15;            // row within 16-wide tile
    const int g = lane >> 4;            // lane half selector

    v8f acc[4][4] = {};

    // A fragment rows (16-bit A 16x32 layout: lane m=r; g=0 -> K 0..7 & 16..23; g=1 -> K 8..15 & 24..31)
    const __bf16* arow[4];
#pragma unroll
    for (int i = 0; i < 4; ++i) arow[i] = A + (size_t)(m0 + 16 * i + r) * K;
    // B fragment rows (16-bit B 32x16 layout: lane n=r; g=0 -> K 0..7 & 16..23; g=1 -> K 8..15 & 24..31) -> W row n, cols k..
    const __bf16* wrow[4];
#pragma unroll
    for (int j = 0; j < 4; ++j) wrow[j] = W + (size_t)(n0 + 16 * j + r) * K;

    const int aoff = 8 * g;
    const int boff = 8 * g;

    for (int k = 0; k < K; k += 32) {
        Frag fa[4], fb[4];
#pragma unroll
        for (int i = 0; i < 4; ++i) {
            fa[i].h[0] = *(const v8bf*)(arow[i] + k + aoff);
            fa[i].h[1] = *(const v8bf*)(arow[i] + k + aoff + 16);
        }
#pragma unroll
        for (int j = 0; j < 4; ++j) {
            fb[j].h[0] = *(const v8bf*)(wrow[j] + k + boff);
            fb[j].h[1] = *(const v8bf*)(wrow[j] + k + boff + 16);
        }
        __builtin_prefetch(arow[0] + k + 128, 0, 3);   // global_prefetch_b8
        __builtin_prefetch(wrow[0] + k + 128, 0, 3);
#pragma unroll
        for (int i = 0; i < 4; ++i)
#pragma unroll
            for (int j = 0; j < 4; ++j)
                acc[i][j] = __builtin_amdgcn_wmma_f32_16x16x32_bf16(
                    /*neg_a=*/false, fa[i].v, /*neg_b=*/false, fb[j].v,
                    /*c_mod=*/(short)0, acc[i][j], /*reuse_a=*/false, /*reuse_b=*/false);
    }

    // Epilogue: C/D 16x16 f32 layout -> row = base + v + 8g, col = base + r
#pragma unroll
    for (int j = 0; j < 4; ++j) {
        const int col = n0 + 16 * j + r;
        const float bv = bias[col];
#pragma unroll
        for (int i = 0; i < 4; ++i) {
            const int mb = m0 + 16 * i + 8 * g;
#pragma unroll
            for (int v = 0; v < 8; ++v)
                C[(size_t)(mb + v) * N + col] = acc[i][j][v] + bv;
        }
    }
}

// ---------------- sequential linear recurrence  h[l] = lam*h[l-1] + u[l] ----------------
// one thread per (b, e) channel; coalesced across e.
__global__ __launch_bounds__(256)
void lru_scan(const float* __restrict__ u,          // [B, L, D] f32
              const float* __restrict__ log_lambda, // [D]
              __bf16* __restrict__ hb,              // [B, L, D] bf16 (for GEMM2)
              float* __restrict__ next_state)       // [B, D] f32
{
    const int idx = blockIdx.x * 256 + threadIdx.x;     // 0 .. B*D-1
    const int e = idx & (D_MODEL - 1);
    const int b = idx >> 11;                            // D_MODEL == 2048
    const float lam = __expf(log_lambda[e]);
    const float* up = u  + (size_t)b * SEQ * D_MODEL + e;
    __bf16*      hp = hb + (size_t)b * SEQ * D_MODEL + e;
    float h = 0.0f;
    for (int l = 0; l < SEQ; ++l) {
        h = fmaf(lam, h, up[(size_t)l * D_MODEL]);
        hp[(size_t)l * D_MODEL] = to_bf16(h);
    }
    next_state[idx] = h;
}

// ---------------- host launcher ----------------
extern "C" void kernel_launch(void* const* d_in, const int* in_sizes, int n_in,
                              void* d_out, int out_size, void* d_ws, size_t ws_size,
                              hipStream_t stream) {
    const float* x          = (const float*)d_in[0];   // [B, L, D]
    const float* log_lambda = (const float*)d_in[1];   // [D]
    const float* W_in       = (const float*)d_in[2];   // [D, D]
    const float* b_in       = (const float*)d_in[3];   // [D]
    const float* W_out      = (const float*)d_in[4];   // [D, D]
    const float* b_out      = (const float*)d_in[5];   // [D]

    float* out        = (float*)d_out;                             // [B, L, D]
    float* next_state = out + (size_t)BATCH * SEQ * D_MODEL;       // [B, D]

    // workspace layout (bytes):
    //   xb  bf16[B*L*D]   @ 0          (64 MiB)
    //   hb  bf16[B*L*D]   @ 64 MiB     (64 MiB)
    //   wib bf16[D*D]     @ 128 MiB    (8 MiB)
    //   wob bf16[D*D]     @ 136 MiB    (8 MiB)
    //   u   f32 [B*L*D]   @ 144 MiB    (128 MiB)      total ~272 MiB
    char* ws = (char*)d_ws;
    __bf16* xb  = (__bf16*)(ws);
    __bf16* hb  = (__bf16*)(ws + (size_t)67108864);
    __bf16* wib = (__bf16*)(ws + (size_t)134217728);
    __bf16* wob = (__bf16*)(ws + (size_t)142606336);
    float*  u   = (float*) (ws + (size_t)150994944);

    const int MX    = BATCH * SEQ;                 // 16384
    const int nX4   = BATCH * SEQ * D_MODEL / 4;   // 8388608
    const int nW4   = D_MODEL * D_MODEL / 4;       // 1048576

    // 1) precision downcast for the matrix path
    cvt_f32_to_bf16<<<(nX4 + 255) / 256, 256, 0, stream>>>(x,     xb,  nX4);
    cvt_f32_to_bf16<<<(nW4 + 255) / 256, 256, 0, stream>>>(W_in,  wib, nW4);
    cvt_f32_to_bf16<<<(nW4 + 255) / 256, 256, 0, stream>>>(W_out, wob, nW4);

    // 2) u = x @ W_in^T + b_in     (f32 accumulate/out)
    dim3 grid(D_MODEL / 256, MX / 128);            // (8, 128)
    gemm_bf16_wmma<<<grid, 256, 0, stream>>>(xb, wib, b_in, u, MX, D_MODEL, D_MODEL);

    // 3) h[l] = lam*h[l-1] + u[l]; emit h as bf16, last state as f32
    lru_scan<<<(BATCH * D_MODEL) / 256, 256, 0, stream>>>(u, log_lambda, hb, next_state);

    // 4) out = h @ W_out^T + b_out (f32 out, straight to d_out)
    gemm_bf16_wmma<<<grid, 256, 0, stream>>>(hb, wob, b_out, out, MX, D_MODEL, D_MODEL);
}
